// SkeletonMambaClassifier_61280593379364
// MI455X (gfx1250) — matrix operand order repaired and checked by reading, not compiled, not run on hardware
//
#include <hip/hip_runtime.h>
#include <hip/hip_bf16.h>

// ---------------------------------------------------------------------------
// SkeletonMambaClassifier for MI455X (gfx1250, wave32, WMMA).
// All GEMMs use v_wmma_f32_16x16x32_f16 (f16 in, f32 accumulate).
// ---------------------------------------------------------------------------

typedef __attribute__((ext_vector_type(16))) _Float16 v16h;
typedef __attribute__((ext_vector_type(8)))  _Float16 v8h;
typedef __attribute__((ext_vector_type(8)))  float    v8f;

#define BB 8
#define TT 1024
#define JJ 25
#define D_MODEL 256
#define D_STATE 16
#define D_CONV 4
#define D_INNER 512
#define DT_RANK 16
#define N_CLASSES 60
#define BT (BB * TT)          // 8192 tokens

enum { ACT_NONE = 0, ACT_SOFTPLUS = 1 };

// ---------------------------------------------------------------------------
// Generic f32 -> f16 weight conversion with zero padding of K to Kpad.
// src: (N, K) row-major f32 ; dst: (N, Kpad) row-major f16.
// ---------------------------------------------------------------------------
__global__ void convert_pad_f16(const float* __restrict__ src, _Float16* __restrict__ dst,
                                int N, int K, int Kpad) {
    int idx = blockIdx.x * blockDim.x + threadIdx.x;
    if (idx >= N * Kpad) return;
    int n = idx / Kpad, k = idx - n * Kpad;
    dst[idx] = (k < K) ? (_Float16)src[(size_t)n * K + k] : (_Float16)0.0f;
}

// ---------------------------------------------------------------------------
// Feature extraction: pos/vel/acc per joint, written as f16 WMMA A operand,
// zero-padded from 225 to 256 columns. Column layout matches the reference's
// concat(...,axis=-1).reshape: col = j*9 + {0..2 pos, 3..5 vel, 6..8 acc}.
// ---------------------------------------------------------------------------
__global__ void build_features(const float* __restrict__ x, _Float16* __restrict__ Ah) {
    int idx = blockIdx.x * blockDim.x + threadIdx.x;     // over BT*JJ
    if (idx >= BT * JJ) return;
    int j  = idx % JJ;
    int bt = idx / JJ;
    int t  = bt % TT;
    const float* xb  = x + (size_t)bt * JJ * 3;          // (b,t)
    const float* xb1 = xb - JJ * 3;                      // (b,t-1) if t>=1
    const float* xb2 = xb - 2 * JJ * 3;                  // (b,t-2) if t>=2
    _Float16* row = Ah + (size_t)bt * 256;
    #pragma unroll
    for (int c = 0; c < 3; ++c) {
        float p   = xb[j * 3 + c] - xb[c];
        float vel = 0.0f, acc = 0.0f;
        if (t >= 1) {
            float p1 = xb1[j * 3 + c] - xb1[c];
            vel = p - p1;
            float vel1 = 0.0f;
            if (t >= 2) {
                float p2 = xb2[j * 3 + c] - xb2[c];
                vel1 = p1 - p2;
            }
            acc = vel - vel1;
        }
        row[j * 9 + c]     = (_Float16)p;
        row[j * 9 + 3 + c] = (_Float16)vel;
        row[j * 9 + 6 + c] = (_Float16)acc;
    }
    if (j == 0) {                                        // zero pad cols 225..255
        for (int k = JJ * 9; k < 256; ++k) row[k] = (_Float16)0.0f;
    }
}

// ---------------------------------------------------------------------------
// WMMA GEMM: C[M,N] (f32) = act(A[M,K]f16 * W[N,K]^T f16 + bias).
// One 16x16 C tile per wave; 8 waves per block (128 rows). K % 32 == 0.
// Fragment gathers follow CDNA5 ISA 7.12.2 VGPR layouts:
//   A (16x32 f16): lanes 0-15 row M=lane, elems = K{0..7,16..23};
//                  lanes 16-31 row M=lane-16, elems = K{8..15,24..31}.
//   B (32x16 f16): lane&15 = column N; lanes <16 K{0..7,16..23}, lanes >=16 K{8..15,24..31}.
//   C/D (16x16 f32): elem r -> (M = r + 8*(lane>=16), N = lane&15).
// ---------------------------------------------------------------------------
__global__ void wmma_gemm_f16f32(const _Float16* __restrict__ A, int lda,
                                 const _Float16* __restrict__ W, int ldw,
                                 float* __restrict__ C, int ldc,
                                 const float* __restrict__ bias,
                                 int M, int K, int act) {
    const int lane = threadIdx.x & 31;
    const int wave = threadIdx.x >> 5;
    const int m0 = (blockIdx.y * 8 + wave) * 16;
    const int n0 = blockIdx.x * 16;
    if (m0 >= M) return;                                  // wave-uniform

    const int hi   = (lane >= 16);
    const int rowA = m0 + (lane & 15);
    const int colB = n0 + (lane & 15);
    const int offA = hi ? 8  : 0;                         // halves
    const int offB = hi ? 8  : 0;                         // halves

    const _Float16* Arow = A + (size_t)rowA * lda;
    const _Float16* Wrow = W + (size_t)colB * ldw;

    v8f c = {};
    for (int k0 = 0; k0 < K; k0 += 32) {
        v8h a_lo = *(const v8h*)(Arow + k0 + offA);
        v8h a_hi = *(const v8h*)(Arow + k0 + offA + 16);
        v8h b_lo = *(const v8h*)(Wrow + k0 + offB);
        v8h b_hi = *(const v8h*)(Wrow + k0 + offB + 16);
        v16h a, b;
        #pragma unroll
        for (int i = 0; i < 8; ++i) {
            a[i] = a_lo[i]; a[i + 8] = a_hi[i];
            b[i] = b_lo[i]; b[i + 8] = b_hi[i];
        }
        c = __builtin_amdgcn_wmma_f32_16x16x32_f16(false, a, false, b,
                                                   (short)0, c, false, false);
    }

    const int mbase = m0 + (hi ? 8 : 0);
    float bv = bias ? bias[colB] : 0.0f;
    #pragma unroll
    for (int r = 0; r < 8; ++r) {
        float v = c[r] + bv;
        if (act == ACT_SOFTPLUS)
            v = (v > 20.0f) ? v : __logf(1.0f + __expf(v));
        C[(size_t)(mbase + r) * ldc + colB] = v;
    }
}

// ---------------------------------------------------------------------------
// LayerNorm (D=256, one token per 256-thread block). Pre-variant -> f16.
// ---------------------------------------------------------------------------
__global__ void layernorm_to_f16(const float* __restrict__ X,
                                 const float* __restrict__ g, const float* __restrict__ b,
                                 _Float16* __restrict__ Yh) {
    __shared__ float red[256];
    const int D = D_MODEL;
    size_t t = blockIdx.x; int tid = threadIdx.x;
    float v = X[t * D + tid];
    red[tid] = v; __syncthreads();
    for (int s = 128; s > 0; s >>= 1) { if (tid < s) red[tid] += red[tid + s]; __syncthreads(); }
    float mu = red[0] / D; __syncthreads();
    float dv = v - mu;
    red[tid] = dv * dv; __syncthreads();
    for (int s = 128; s > 0; s >>= 1) { if (tid < s) red[tid] += red[tid + s]; __syncthreads(); }
    float var = red[0] / D;
    float y = dv * rsqrtf(var + 1e-5f) * g[tid] + b[tid];
    Yh[t * D + tid] = (_Float16)y;
}

// Post-variant: residual add then LN, f32 out.
__global__ void residual_layernorm(const float* __restrict__ X, const float* __restrict__ R,
                                   const float* __restrict__ g, const float* __restrict__ b,
                                   float* __restrict__ Y) {
    __shared__ float red[256];
    const int D = D_MODEL;
    size_t t = blockIdx.x; int tid = threadIdx.x;
    float v = X[t * D + tid] + R[t * D + tid];
    red[tid] = v; __syncthreads();
    for (int s = 128; s > 0; s >>= 1) { if (tid < s) red[tid] += red[tid + s]; __syncthreads(); }
    float mu = red[0] / D; __syncthreads();
    float dv = v - mu;
    red[tid] = dv * dv; __syncthreads();
    for (int s = 128; s > 0; s >>= 1) { if (tid < s) red[tid] += red[tid + s]; __syncthreads(); }
    float var = red[0] / D;
    Y[t * D + tid] = dv * rsqrtf(var + 1e-5f) * g[tid] + b[tid];
}

// ---------------------------------------------------------------------------
// Depthwise causal conv (kernel 4, left pad 3) + bias + SiLU.
// xm = first 512 cols of xz (row stride 1024). Emits f32 (scan u) + f16 (GEMM A).
// ---------------------------------------------------------------------------
__global__ void dwconv_silu(const float* __restrict__ xz,
                            const float* __restrict__ cw, const float* __restrict__ cb,
                            float* __restrict__ xmc, _Float16* __restrict__ xmch) {
    int idx = blockIdx.x * blockDim.x + threadIdx.x;      // over BT*512
    if (idx >= BT * D_INNER) return;
    int d  = idx % D_INNER;
    int bt = idx / D_INNER;
    int t  = bt % TT;
    int b  = bt / TT;
    float acc = cb[d];
    #pragma unroll
    for (int i = 0; i < D_CONV; ++i) {
        int tt2 = t - (D_CONV - 1) + i;
        if (tt2 >= 0)
            acc += cw[d * D_CONV + i] * xz[((size_t)(b * TT + tt2)) * 1024 + d];
    }
    float s = acc * (1.0f / (1.0f + __expf(-acc)));
    xmc[idx]  = s;
    xmch[idx] = (_Float16)s;
}

// dt columns (0..15 of dbl) -> f16 padded to 32 for the dt_proj WMMA GEMM.
__global__ void extract_dt(const float* __restrict__ dbl, _Float16* __restrict__ dth) {
    int idx = blockIdx.x * blockDim.x + threadIdx.x;      // over BT*32
    if (idx >= BT * 32) return;
    int k = idx % 32, row = idx / 32;
    dth[idx] = (k < DT_RANK) ? (_Float16)dbl[(size_t)row * 48 + k] : (_Float16)0.0f;
}

// ---------------------------------------------------------------------------
// Selective scan: one thread per (b, d) channel, h[16] in registers,
// sequential over T. y = (sum_n h_n * C_n + u*D) * silu(res). Emits f16 y
// (A operand of the out_proj WMMA GEMM).
// ---------------------------------------------------------------------------
__global__ void selective_scan_kernel(const float* __restrict__ delta,
                                      const float* __restrict__ u,
                                      const float* __restrict__ dbl,
                                      const float* __restrict__ xz,
                                      const float* __restrict__ A_log,
                                      const float* __restrict__ Dp,
                                      _Float16* __restrict__ yh) {
    int d = (blockIdx.x & 1) * 256 + threadIdx.x;         // 0..511
    int b = blockIdx.x >> 1;                              // 0..7
    float Arow[D_STATE], h[D_STATE];
    #pragma unroll
    for (int n = 0; n < D_STATE; ++n) {
        Arow[n] = -__expf(A_log[d * D_STATE + n]);
        h[n] = 0.0f;
    }
    const float Dd = Dp[d];
    for (int t = 0; t < TT; ++t) {
        size_t bt = (size_t)b * TT + t;
        float dl = delta[bt * D_INNER + d];
        float uu = u[bt * D_INNER + d];
        const float* dr = dbl + bt * 48;                  // [dt(16) | B(16) | C(16)]
        float y = 0.0f;
        #pragma unroll
        for (int n = 0; n < D_STATE; ++n) {
            float dA = __expf(dl * Arow[n]);
            h[n] = dA * h[n] + dl * dr[16 + n] * uu;      // broadcast loads
            y += h[n] * dr[32 + n];
        }
        y += uu * Dd;
        float r = xz[bt * 1024 + D_INNER + d];            // res
        y *= r * (1.0f / (1.0f + __expf(-r)));            // * silu(res)
        yh[bt * D_INNER + d] = (_Float16)y;
    }
}

// Mean over T: pooled[b][d] = (1/T) sum_t feat2[b,t,d].
__global__ void mean_over_t(const float* __restrict__ X, float* __restrict__ pooled) {
    int b = blockIdx.x, d = threadIdx.x;                  // 8 blocks x 256
    float s = 0.0f;
    for (int t = 0; t < TT; ++t) s += X[((size_t)b * TT + t) * D_MODEL + d];
    pooled[b * D_MODEL + d] = s * (1.0f / TT);
}

// Final classifier: out[b][c] = pooled[b,:] . fine_w[c,:] + fine_b[c].
__global__ void classifier(const float* __restrict__ pooled,
                           const float* __restrict__ fw, const float* __restrict__ fb,
                           float* __restrict__ out) {
    int idx = blockIdx.x * blockDim.x + threadIdx.x;
    if (idx >= BB * N_CLASSES) return;
    int b = idx / N_CLASSES, c = idx % N_CLASSES;
    float s = fb[c];
    for (int k = 0; k < D_MODEL; ++k) s += pooled[b * D_MODEL + k] * fw[c * D_MODEL + k];
    out[idx] = s;
}

// ---------------------------------------------------------------------------
extern "C" void kernel_launch(void* const* d_in, const int* in_sizes, int n_in,
                              void* d_out, int out_size, void* d_ws, size_t ws_size,
                              hipStream_t stream) {
    const float* x        = (const float*)d_in[0];
    const float* embed_w  = (const float*)d_in[1];
    const float* embed_b  = (const float*)d_in[2];
    const float* pre_g    = (const float*)d_in[3];
    const float* pre_b    = (const float*)d_in[4];
    const float* inproj_w = (const float*)d_in[5];
    const float* conv_w   = (const float*)d_in[6];
    const float* conv_b   = (const float*)d_in[7];
    const float* xproj_w  = (const float*)d_in[8];
    const float* dtproj_w = (const float*)d_in[9];
    const float* dtproj_b = (const float*)d_in[10];
    const float* A_log    = (const float*)d_in[11];
    const float* Dparam   = (const float*)d_in[12];
    const float* outproj_w= (const float*)d_in[13];
    const float* post_g   = (const float*)d_in[14];
    const float* post_b   = (const float*)d_in[15];
    const float* fine_w   = (const float*)d_in[16];
    const float* fine_b   = (const float*)d_in[17];
    float* out = (float*)d_out;

    char* ws = (char*)d_ws;
    size_t off = 0;
    auto alloc = [&](size_t bytes) -> char* {
        char* p = ws + off;
        off = (off + bytes + 255) & ~(size_t)255;
        return p;
    };

    // f16 weights (K padded to multiple of 32)
    _Float16* wh_embed  = (_Float16*)alloc((size_t)256  * 256 * 2);
    _Float16* wh_inproj = (_Float16*)alloc((size_t)1024 * 256 * 2);
    _Float16* wh_xproj  = (_Float16*)alloc((size_t)48   * 512 * 2);
    _Float16* wh_dtproj = (_Float16*)alloc((size_t)512  * 32  * 2);
    _Float16* wh_outproj= (_Float16*)alloc((size_t)256  * 512 * 2);
    // activations
    _Float16* Afeat_h = (_Float16*)alloc((size_t)BT * 256 * 2);
    float*    featF32 = (float*)   alloc((size_t)BT * 256 * 4);
    _Float16* hn_h    = (_Float16*)alloc((size_t)BT * 256 * 2);
    float*    xz      = (float*)   alloc((size_t)BT * 1024 * 4);
    float*    xmc     = (float*)   alloc((size_t)BT * 512 * 4);
    _Float16* xmc_h   = (_Float16*)alloc((size_t)BT * 512 * 2);
    float*    dbl     = (float*)   alloc((size_t)BT * 48 * 4);
    _Float16* dt_h    = (_Float16*)alloc((size_t)BT * 32 * 2);
    float*    delta   = (float*)   alloc((size_t)BT * 512 * 4);
    _Float16* y_h     = (_Float16*)alloc((size_t)BT * 512 * 2);
    float*    y2      = (float*)   alloc((size_t)BT * 256 * 4);
    float*    feat2   = (float*)   alloc((size_t)BT * 256 * 4);
    float*    pooled  = (float*)   alloc((size_t)BB * 256 * 4);

    const dim3 blk(256);
    auto cdiv = [](int a, int b) { return (a + b - 1) / b; };

    // 1) weight conversion/padding
    convert_pad_f16<<<cdiv(256 * 256, 256),  blk, 0, stream>>>(embed_w,   wh_embed,   256,  225, 256);
    convert_pad_f16<<<cdiv(1024 * 256, 256), blk, 0, stream>>>(inproj_w,  wh_inproj,  1024, 256, 256);
    convert_pad_f16<<<cdiv(48 * 512, 256),   blk, 0, stream>>>(xproj_w,   wh_xproj,   48,   512, 512);
    convert_pad_f16<<<cdiv(512 * 32, 256),   blk, 0, stream>>>(dtproj_w,  wh_dtproj,  512,  16,  32);
    convert_pad_f16<<<cdiv(256 * 512, 256),  blk, 0, stream>>>(outproj_w, wh_outproj, 256,  512, 512);

    // 2) features -> f16 WMMA A operand
    build_features<<<cdiv(BT * JJ, 256), blk, 0, stream>>>(x, Afeat_h);

    // 3) embed GEMM: (8192x256) = feat(8192x256) * embed_w^T, + bias
    wmma_gemm_f16f32<<<dim3(256 / 16, BT / 128), blk, 0, stream>>>(
        Afeat_h, 256, wh_embed, 256, featF32, 256, embed_b, BT, 256, ACT_NONE);

    // 4) pre-LN -> f16
    layernorm_to_f16<<<BT, blk, 0, stream>>>(featF32, pre_g, pre_b, hn_h);

    // 5) in_proj GEMM: (8192x1024)
    wmma_gemm_f16f32<<<dim3(1024 / 16, BT / 128), blk, 0, stream>>>(
        hn_h, 256, wh_inproj, 256, xz, 1024, nullptr, BT, 256, ACT_NONE);

    // 6) depthwise causal conv + SiLU
    dwconv_silu<<<cdiv(BT * 512, 256), blk, 0, stream>>>(xz, conv_w, conv_b, xmc, xmc_h);

    // 7) x_proj GEMM: (8192x48) = xm(8192x512) * x_proj_w^T
    wmma_gemm_f16f32<<<dim3(48 / 16, BT / 128), blk, 0, stream>>>(
        xmc_h, 512, wh_xproj, 512, dbl, 48, nullptr, BT, 512, ACT_NONE);

    // 8) dt slice -> padded f16
    extract_dt<<<cdiv(BT * 32, 256), blk, 0, stream>>>(dbl, dt_h);

    // 9) dt_proj GEMM + softplus -> delta (8192x512)
    wmma_gemm_f16f32<<<dim3(512 / 16, BT / 128), blk, 0, stream>>>(
        dt_h, 32, wh_dtproj, 32, delta, 512, dtproj_b, BT, 32, ACT_SOFTPLUS);

    // 10) selective scan (sequential over T, 4096 channels in parallel)
    selective_scan_kernel<<<BB * 2, blk, 0, stream>>>(delta, xmc, dbl, xz, A_log, Dparam, y_h);

    // 11) out_proj GEMM: (8192x256) = y(8192x512) * out_proj_w^T
    wmma_gemm_f16f32<<<dim3(256 / 16, BT / 128), blk, 0, stream>>>(
        y_h, 512, wh_outproj, 512, y2, 256, nullptr, BT, 512, ACT_NONE);

    // 12) residual + post-LN
    residual_layernorm<<<BT, blk, 0, stream>>>(featF32, y2, post_g, post_b, feat2);

    // 13) temporal mean pool
    mean_over_t<<<BB, blk, 0, stream>>>(feat2, pooled);

    // 14) classifier head
    classifier<<<cdiv(BB * N_CLASSES, 256), blk, 0, stream>>>(pooled, fine_w, fine_b, out);
}
